// AFTLocal_1451698946442
// MI455X (gfx1250) — compile-verified, matrix operand order checked by reading, not run on hardware
//
#include <hip/hip_runtime.h>
#include <hip/hip_bf16.h>

typedef __attribute__((ext_vector_type(16))) __bf16 v16bf;
typedef __attribute__((ext_vector_type(8)))  float  v8f;

#define BATCH 8
#define TSEQ  2048
#define DIMSZ 1024
#define HIDSZ 64
#define BT    (BATCH * TSEQ)

// ---------- workspace layout (bytes) ----------
#define WQKVT_OFF 0u               // bf16 [192][1024]      = 393216 B
#define WPT_OFF   393216u          // bf16 [1024][64]       = 131072 B
#define EW_OFF    524288u          // bf16 [2048][2048]     = 8388608 B
#define QF_OFF    8912896u         // f32  [BT][64]         = 4194304 B
#define KT_OFF    13107200u        // bf16 [B*64][T] (K^T)  = 2097152 B
#define KVT_OFF   15204352u        // bf16 [B*64][T] (KV^T) = 2097152 B
#define NUM_OFF   17301504u        // f32  [BT][64]         = 4194304 B
#define DEN_OFF   21495808u        // f32  [BT][64]         = 4194304 B
#define YT_OFF    25690112u        // bf16 [BT][64]         = 2097152 B

__device__ __forceinline__ unsigned short f2bf(float f) {
  unsigned int u = __float_as_uint(f);
  u += 0x7FFFu + ((u >> 16) & 1u);      // round-to-nearest-even
  return (unsigned short)(u >> 16);
}

union FragBF { v16bf v; unsigned short u[16]; };

// 16x32 bf16 fragment: lane = n0 + lane%16 along leading dim, k contiguous.
// ISA 7.12.2: halves[0..7] = k0+half*8+0..7, halves[8..15] = k0+16+half*8+0..7.
__device__ __forceinline__ v16bf load_frag_g(const unsigned short* __restrict__ base,
                                             int n0, int ld, int k0, int lane) {
  int n = n0 + (lane & 15);
  int half = lane >> 4;
  const unsigned short* p = base + (size_t)n * ld + k0 + half * 8;
  FragBF f;
#pragma unroll
  for (int i = 0; i < 8; ++i) { f.u[i] = p[i]; f.u[8 + i] = p[16 + i]; }
  return f.v;
}

#define WMMA_BF16(A, B, C) \
  __builtin_amdgcn_wmma_f32_16x16x32_bf16(false, (A), false, (B), (short)0, (C), false, false)

// ---------------- kernel 1: weight conversion / transpose ----------------
__global__ void k_prep_w(const float* __restrict__ Wq, const float* __restrict__ Wk,
                         const float* __restrict__ Wv, const float* __restrict__ Wp,
                         unsigned short* __restrict__ Wqkv_t, unsigned short* __restrict__ Wpt) {
  int idx = blockIdx.x * blockDim.x + threadIdx.x;
  if (idx < 192 * 1024) {
    int c = idx >> 10;            // projection column (row of Wqkv_t)
    int k = idx & 1023;
    float v;
    if (c < 64)       v = Wq[k * 64 + c];
    else if (c < 128) v = Wk[k * 64 + (c - 64)];
    else              v = Wv[k * 64 + (c - 128)];
    Wqkv_t[idx] = f2bf(v);
  } else {
    int j = idx - 192 * 1024;
    if (j < 1024 * 64) {
      int n = j >> 6;
      int k = j & 63;
      Wpt[j] = f2bf(Wp[k * 1024 + n]);
    }
  }
}

// ---------------- kernel 2: ew = exp(band ? wbias : 0) -> bf16 ----------------
__global__ void k_ew(const float* __restrict__ wbias, const int* __restrict__ sptr,
                     unsigned short* __restrict__ ew) {
  int idx = blockIdx.x * blockDim.x + threadIdx.x;     // [0, 2048*2048)
  int t = idx >> 11, si = idx & 2047;
  int d = t - si; if (d < 0) d = -d;
  float w = (d < sptr[0]) ? wbias[idx] : 0.0f;
  ew[idx] = f2bf(__expf(w));
}

// ---------------- kernel 3: fused QKV projection + activations ----------------
// block = 256 threads (8 waves), 64 rows x 192 cols. Wave w: row subtile
// r=w&3, col half c=w>>2 owns Q/K/V cols [32c,32c+32). Epilogue writes K and
// K*V TRANSPOSED ([b*64+h][t], t contiguous -> b128 stores) for k_aft.
__global__ __launch_bounds__(256) void k_proj(
    const float* __restrict__ x, const unsigned short* __restrict__ Wqkv_t,
    const float* __restrict__ bq, const float* __restrict__ bk, const float* __restrict__ bv,
    float* __restrict__ Qf, unsigned short* __restrict__ Kt, unsigned short* __restrict__ KVt) {
  __shared__ unsigned short shA[64 * 32];          // 4 KB bf16 x-tile
  int tid = threadIdx.x, lane = tid & 31, w = tid >> 5;
  int r = w & 3, c = w >> 2;
  int row0 = blockIdx.x * 64;

  v8f acc[6] = {};                                 // [0,1]=Q [2,3]=K [4,5]=V

  for (int k0 = 0; k0 < DIMSZ; k0 += 32) {
    {   // stage x[row0..row0+64, k0..k0+32) as bf16
      int rr = tid >> 2;
      int cc = (tid & 3) * 8;
      const float4* px = (const float4*)(x + (size_t)(row0 + rr) * DIMSZ + k0 + cc);
      float4 a = px[0], b = px[1];
      unsigned short* d = &shA[rr * 32 + cc];
      d[0] = f2bf(a.x); d[1] = f2bf(a.y); d[2] = f2bf(a.z); d[3] = f2bf(a.w);
      d[4] = f2bf(b.x); d[5] = f2bf(b.y); d[6] = f2bf(b.z); d[7] = f2bf(b.w);
    }
    __syncthreads();
    v16bf afrag;
    {   // A fragment from LDS (rows r*16 .. r*16+16)
      int m = lane & 15, half = lane >> 4;
      const unsigned short* p = &shA[(r * 16 + m) * 32 + half * 8];
      FragBF f;
#pragma unroll
      for (int i = 0; i < 8; ++i) { f.u[i] = p[i]; f.u[8 + i] = p[16 + i]; }
      afrag = f.v;
    }
#pragma unroll
    for (int j = 0; j < 2; ++j) {
      int colQ = 32 * c + 16 * j;
      v16bf fq = load_frag_g(Wqkv_t, colQ,        DIMSZ, k0, lane);
      v16bf fk = load_frag_g(Wqkv_t, 64  + colQ,  DIMSZ, k0, lane);
      v16bf fv = load_frag_g(Wqkv_t, 128 + colQ,  DIMSZ, k0, lane);
      acc[j]     = WMMA_BF16(afrag, fq, acc[j]);
      acc[2 + j] = WMMA_BF16(afrag, fk, acc[2 + j]);
      acc[4 + j] = WMMA_BF16(afrag, fv, acc[4 + j]);
    }
    __syncthreads();
  }

  int m = lane & 15, half = lane >> 4;
  int b   = row0 >> 11;                     // batch (row0 multiple of 64, T=2048)
  int tb  = (row0 & (TSEQ - 1)) + r * 16 + half * 8;   // first t of this lane's 8 rows
#pragma unroll
  for (int j = 0; j < 2; ++j) {
    int h = 32 * c + 16 * j + m;
    float bqv = bq[h], bkv = bk[h], bvv = bv[h];
    unsigned short tk[8], tkv[8];
#pragma unroll
    for (int d = 0; d < 8; ++d) {
      int row = row0 + r * 16 + half * 8 + d;
      float q = acc[j][d] + bqv;
      Qf[(size_t)row * HIDSZ + h] = 1.0f / (1.0f + __expf(-q));   // sigmoid
      float kv = __expf(acc[2 + j][d] + bkv);                     // exp(K)
      float vv = acc[4 + j][d] + bvv;
      tk[d]  = f2bf(kv);
      tkv[d] = f2bf(kv * vv);
    }
    size_t to = ((size_t)b * HIDSZ + h) * TSEQ + tb;    // 16B-aligned, t-contig
    *(uint4*)(Kt  + to) = *(const uint4*)tk;
    *(uint4*)(KVt + to) = *(const uint4*)tkv;
  }
}

// ---------------- kernel 4: [num|den] = ew @ [KV|K] per batch ----------------
// grid (T/64, B); block 256 = 8 waves; 64 t-rows x 128 cols per block.
// Pure global->WMMA loop: no LDS, no barriers (operands are L2-resident).
// Wave w: row-subtile pair rp=(w&1)*2, col-subtile pair cp=(w>>1)*2.
// Col subtiles 0..3 -> num (KV^T), 4..7 -> den (K^T).
__global__ __launch_bounds__(256) void k_aft(
    const unsigned short* __restrict__ ew, const unsigned short* __restrict__ Kt,
    const unsigned short* __restrict__ KVt,
    float* __restrict__ num, float* __restrict__ den) {
  int tid = threadIdx.x, lane = tid & 31, w = tid >> 5;
  int rp = (w & 1) * 2;
  int cp = (w >> 1) * 2;
  int t0 = blockIdx.x * 64;
  int b  = blockIdx.y;
  const unsigned short* Bb[2];
  int h0[2];
#pragma unroll
  for (int b2 = 0; b2 < 2; ++b2) {
    int cs = cp + b2;
    Bb[b2] = ((cs < 4) ? KVt : Kt) + (size_t)b * HIDSZ * TSEQ;
    h0[b2] = (cs < 4) ? 16 * cs : 16 * (cs - 4);
  }

  v8f acc[4] = {};                  // [a*2+b2]: rows rp+a, cols cp+b2
#pragma unroll 2
  for (int s0 = 0; s0 < TSEQ; s0 += 32) {
    // prefetch next ew chunk for this lane's row (gfx1250 global_prefetch_b8)
    __builtin_prefetch(ew + (size_t)(t0 + rp * 16 + (lane & 15)) * TSEQ + s0 + 128, 0, 3);
    v16bf a0 = load_frag_g(ew, t0 + rp * 16,       TSEQ, s0, lane);
    v16bf a1 = load_frag_g(ew, t0 + (rp + 1) * 16, TSEQ, s0, lane);
#pragma unroll
    for (int b2 = 0; b2 < 2; ++b2) {
      v16bf bf = load_frag_g(Bb[b2], h0[b2], TSEQ, s0, lane);
      acc[0 * 2 + b2] = WMMA_BF16(a0, bf, acc[0 * 2 + b2]);
      acc[1 * 2 + b2] = WMMA_BF16(a1, bf, acc[1 * 2 + b2]);
    }
  }

  int n = lane & 15, half = lane >> 4;
#pragma unroll
  for (int a = 0; a < 2; ++a) {
#pragma unroll
    for (int b2 = 0; b2 < 2; ++b2) {
      int cs = cp + b2;
      float* dst = (cs < 4) ? num : den;
      int h = h0[b2] + n;
#pragma unroll
      for (int d = 0; d < 8; ++d) {
        int row = t0 + (rp + a) * 16 + half * 8 + d;
        dst[((size_t)b * TSEQ + row) * HIDSZ + h] = acc[a * 2 + b2][d];
      }
    }
  }
}

// ---------------- kernel 5: Yt = Q * num / den -> bf16 ----------------
__global__ void k_yt(const float* __restrict__ Qf, const float* __restrict__ num,
                     const float* __restrict__ den, unsigned short* __restrict__ Ytbf) {
  int idx = blockIdx.x * blockDim.x + threadIdx.x;
  Ytbf[idx] = f2bf(Qf[idx] * num[idx] / den[idx]);
}

// ---------------- kernel 6: out = Yt @ Wp + bp ----------------
// grid (BT/64, 1024/128); block 256 = 8 waves; 64 rows x 128 cols; K = 64.
__global__ __launch_bounds__(256) void k_outproj(
    const unsigned short* __restrict__ Ytbf, const unsigned short* __restrict__ Wpt,
    const float* __restrict__ bp, float* __restrict__ out) {
  int tid = threadIdx.x, lane = tid & 31, w = tid >> 5;
  int r = w & 3, c = w >> 2;
  int row0 = blockIdx.x * 64;
  int col0 = blockIdx.y * 128 + c * 64;
  v8f acc[4] = {};
#pragma unroll
  for (int k0 = 0; k0 < HIDSZ; k0 += 32) {
    v16bf afrag = load_frag_g(Ytbf, row0 + r * 16, HIDSZ, k0, lane);
#pragma unroll
    for (int j = 0; j < 4; ++j) {
      v16bf bfrag = load_frag_g(Wpt, col0 + 16 * j, HIDSZ, k0, lane);
      acc[j] = WMMA_BF16(afrag, bfrag, acc[j]);
    }
  }
  int n = lane & 15, half = lane >> 4;
#pragma unroll
  for (int j = 0; j < 4; ++j) {
    int col = col0 + 16 * j + n;
    float bias = bp[col];
#pragma unroll
    for (int d = 0; d < 8; ++d) {
      int row = row0 + r * 16 + half * 8 + d;
      out[(size_t)row * DIMSZ + col] = acc[j][d] + bias;
    }
  }
}

extern "C" void kernel_launch(void* const* d_in, const int* in_sizes, int n_in,
                              void* d_out, int out_size, void* d_ws, size_t ws_size,
                              hipStream_t stream) {
  const float* x     = (const float*)d_in[0];
  const float* Wq    = (const float*)d_in[1];
  const float* bq    = (const float*)d_in[2];
  const float* Wk    = (const float*)d_in[3];
  const float* bk    = (const float*)d_in[4];
  const float* Wv    = (const float*)d_in[5];
  const float* bv    = (const float*)d_in[6];
  const float* Wp    = (const float*)d_in[7];
  const float* bp    = (const float*)d_in[8];
  const float* wbias = (const float*)d_in[9];
  const int*   sptr  = (const int*)d_in[10];
  float* out = (float*)d_out;

  char* ws = (char*)d_ws;
  unsigned short* Wqkv_t = (unsigned short*)(ws + WQKVT_OFF);
  unsigned short* Wpt    = (unsigned short*)(ws + WPT_OFF);
  unsigned short* ew     = (unsigned short*)(ws + EW_OFF);
  float*          Qf     = (float*)(ws + QF_OFF);
  unsigned short* Kt     = (unsigned short*)(ws + KT_OFF);
  unsigned short* KVt    = (unsigned short*)(ws + KVT_OFF);
  float*          num    = (float*)(ws + NUM_OFF);
  float*          den    = (float*)(ws + DEN_OFF);
  unsigned short* Ytbf   = (unsigned short*)(ws + YT_OFF);

  k_prep_w<<<(192 * 1024 + 64 * 1024) / 256, 256, 0, stream>>>(Wq, Wk, Wv, Wp, Wqkv_t, Wpt);
  k_ew<<<(TSEQ * TSEQ) / 256, 256, 0, stream>>>(wbias, sptr, ew);
  k_proj<<<BT / 64, 256, 0, stream>>>(x, Wqkv_t, bq, bk, bv, Qf, Kt, KVt);
  k_aft<<<dim3(TSEQ / 64, BATCH), 256, 0, stream>>>(ew, Kt, KVt, num, den);
  k_yt<<<(BT * HIDSZ) / 256, 256, 0, stream>>>(Qf, num, den, Ytbf);
  k_outproj<<<dim3(BT / 64, DIMSZ / 128), 256, 0, stream>>>(Ytbf, Wpt, bp, out);
}
